// GRU_18468359372937
// MI455X (gfx1250) — compile-verified, matrix operand order checked by reading, not run on hardware
//
#include <hip/hip_runtime.h>

// ---------------- problem sizes ----------------
#define VOCAB 32000
#define EMB   512
#define HID   1024
#define TT    256
#define BB    16
#define MROWS (TT*BB)          // 4096
#define G3    (3*HID)          // 3072

typedef __attribute__((ext_vector_type(16))) __bf16 v16bf;
typedef __attribute__((ext_vector_type(8)))  float  v8f;

__device__ __forceinline__ v8f wmma_bf16(v16bf a, v16bf b, v8f c) {
  // D = A(16x32) * B(32x16) + C, fp32 accumulate
  return __builtin_amdgcn_wmma_f32_16x16x32_bf16(
      false, a, false, b, (short)0, c, false, false);
}

// ---------------- TDM support detection ----------------
#if defined(__has_builtin)
#if __has_builtin(__builtin_amdgcn_tensor_load_to_lds)
#define USE_TDM 1
#endif
#endif
#ifndef USE_TDM
#define USE_TDM 0
#endif

#if USE_TDM
typedef __attribute__((ext_vector_type(4))) unsigned int v4u_t;
typedef __attribute__((ext_vector_type(8))) int          v8i_t;
typedef __attribute__((ext_vector_type(4))) int          v4i_t;

// Issue a TDM load of a 2D tile [tile1 rows x tile0 elems] of bf16 from a
// row-major tensor [tdim1][tdim0] (row stride = stride0 elems) into LDS.
// D# layout per CDNA5 ISA ch.8 (group0 128b, group1 256b; groups2/3 zero).
// This toolchain exposes the 6-arg builtin (extra int32x8 group + cpol).
__device__ __forceinline__ void tdm_load_2d_bf16(unsigned int lds_off,
                                                 const void* gaddr,
                                                 unsigned int tdim0,
                                                 unsigned int tdim1,
                                                 unsigned int stride0,
                                                 unsigned int tile0,
                                                 unsigned int tile1) {
  unsigned long long ga = (unsigned long long)(uintptr_t)gaddr;
  v4u_t g0;
  g0[0] = 1u;                                    // count=1, load, no gather
  g0[1] = lds_off;                               // lds_addr (bytes)
  g0[2] = (unsigned int)ga;                      // global_addr[31:0]
  g0[3] = (unsigned int)((ga >> 32) & 0x1FFFFFFu) | (2u << 30);  // type=2
  v8i_t g1;
  g1[0] = (int)(1u << 16);                       // data_size=1 -> 2 bytes
  g1[1] = (int)((tdim0 & 0xFFFFu) << 16);        // tensor_dim0[15:0]
  g1[2] = (int)(((tdim0 >> 16) & 0xFFFFu) | ((tdim1 & 0xFFFFu) << 16));
  g1[3] = (int)(((tdim1 >> 16) & 0xFFFFu) | (tile0 << 16));  // tile_dim0
  g1[4] = (int)tile1;                            // tile_dim1 (tile_dim2=0)
  g1[5] = (int)stride0;                          // tensor_dim0_stride[31:0]
  g1[6] = 0;                                     // stride0 hi, dim1_stride lo
  g1[7] = 0;
  v4i_t z4 = {0, 0, 0, 0};
  v8i_t z8 = {0, 0, 0, 0, 0, 0, 0, 0};
  __builtin_amdgcn_tensor_load_to_lds(g0, g1, z4, z4, z8, 0);
}
#endif

__device__ __forceinline__ void wait_tensor_le2() {
#if defined(__has_builtin) && __has_builtin(__builtin_amdgcn_s_wait_tensorcnt)
  __builtin_amdgcn_s_wait_tensorcnt(2);
#else
  asm volatile("s_wait_tensorcnt 0x2" ::: "memory");
#endif
}
__device__ __forceinline__ void wait_tensor_le0() {
#if defined(__has_builtin) && __has_builtin(__builtin_amdgcn_s_wait_tensorcnt)
  __builtin_amdgcn_s_wait_tensorcnt(0);
#else
  asm volatile("s_wait_tensorcnt 0x0" ::: "memory");
#endif
}

// ---------------- workspace layout (bytes) ----------------
#define OFF_XE    ((size_t)0)                       // [4096][512]  bf16   4 MB
#define OFF_WHH   ((size_t)4194304)                 // [3][H][H]    bf16   6 MB
#define OFF_WXX   ((size_t)10485760)                // [3H][E]      bf16   3 MB
#define OFF_WD    ((size_t)13631488)                // [V][H]       bf16  64 MB
#define OFF_GX    ((size_t)79167488)                // [4096][3H]   f32   48 MB
#define OFF_HSEQ  ((size_t)129499136)               // [4096][H]    bf16   8 MB
#define OFF_HF    ((size_t)137887744)               // [B][H]       f32
#define OFF_HB    ((size_t)137953280)               // [B][H]       bf16
#define OFF_RH    ((size_t)137986048)               // [B][H]       bf16
#define OFF_BAR   ((size_t)138018816)               // barrier counter

// ---------------- init: h state + barrier ----------------
__global__ void k_init(const float* h0, float* hf, __bf16* hb, int* bar) {
  int i = blockIdx.x * blockDim.x + threadIdx.x;
  if (i == 0) *bar = 0;
  for (int j = i; j < BB * HID; j += gridDim.x * blockDim.x) {
    float v = h0[j];
    hf[j] = v;
    hb[j] = (__bf16)v;
  }
}

// ---------------- embedding gather -> bf16 ----------------
__global__ void k_gather(const int* x, const float* emb, __bf16* xe) {
  int i = blockIdx.x * blockDim.x + threadIdx.x;
  if (i >= MROWS * EMB) return;
  int tb = i / EMB, e = i % EMB;
  xe[i] = (__bf16)emb[(size_t)x[tb] * EMB + e];
}

// ---------------- weight converters ----------------
__global__ void k_conv_wh(const float* Wz, const float* Wr, const float* W,
                          __bf16* Whh) {
  int i = blockIdx.x * blockDim.x + threadIdx.x;
  if (i >= 3 * HID * HID) return;
  int g = i / (HID * HID), rem = i % (HID * HID);
  int n = rem / HID, k = rem % HID;
  const float* Wg = (g == 0) ? Wz : (g == 1) ? Wr : W;
  Whh[i] = (__bf16)Wg[(size_t)n * (EMB + HID) + k];      // h-part columns
}

__global__ void k_conv_wx(const float* Wz, const float* Wr, const float* W,
                          __bf16* Wxx) {
  int i = blockIdx.x * blockDim.x + threadIdx.x;
  if (i >= 3 * HID * EMB) return;
  int g = i / (HID * EMB), rem = i % (HID * EMB);
  int n = rem / EMB, k = rem % EMB;
  const float* Wg = (g == 0) ? Wz : (g == 1) ? Wr : W;
  Wxx[i] = (__bf16)Wg[(size_t)n * (EMB + HID) + HID + k]; // x-part columns
}

__global__ void k_conv_wd(const float* Wd, __bf16* Wdb) {
  int i = blockIdx.x * blockDim.x + threadIdx.x;
  if (i >= VOCAB * HID) return;
  Wdb[i] = (__bf16)Wd[i];
}

// ---------------- input projection GEMM: Gx = xe @ Wxx^T ----------------
__global__ void __launch_bounds__(256) k_gx_gemm(const __bf16* xe,
                                                 const __bf16* Wxx,
                                                 float* Gx) {
  int lane = threadIdx.x & 31, wv = threadIdx.x >> 5;
  int n0 = (blockIdx.x * 8 + wv) * 16;
  int m0 = blockIdx.y * 16;
  int row = lane & 15, khalf = (lane >> 4) * 16;
  v8f acc = {};
  for (int k0 = 0; k0 < EMB; k0 += 32) {
    v16bf a = *(const v16bf*)&xe[(size_t)(m0 + row) * EMB + k0 + khalf];
    v16bf b = *(const v16bf*)&Wxx[(size_t)(n0 + row) * EMB + k0 + khalf];
    acc = wmma_bf16(a, b, acc);
  }
  int mb = m0 + (lane >> 4) * 8, nb = n0 + (lane & 15);
  for (int i = 0; i < 8; ++i) Gx[(size_t)(mb + i) * G3 + nb] = acc[i];
}

// ---------------- grid barrier ----------------
__device__ __forceinline__ void gbar(int* bar, int target) {
  __syncthreads();
  if (threadIdx.x == 0) {
    __threadfence();
    __hip_atomic_fetch_add(bar, 1, __ATOMIC_RELEASE, __HIP_MEMORY_SCOPE_AGENT);
    while (__hip_atomic_load(bar, __ATOMIC_ACQUIRE,
                             __HIP_MEMORY_SCOPE_AGENT) < target) {
    }
  }
  __syncthreads();
}

// ---------------- persistent GRU recurrence ----------------
// 64 blocks, each owns a 16-wide slice of HID. 4 waves/block.
// Activation panels (h, r*h) are DMAed into LDS by the TDM each phase.
__global__ void __launch_bounds__(128) k_gru_rec(
    const __bf16* Whh, const float* Gx,
    float* hf, __bf16* hb, __bf16* rh,
    float* out_h, __bf16* hseq, int* bar) {
  __shared__ __bf16 a_sh[BB][HID];       // 32 KB A-matrix staging (h or r*h)
  __shared__ float  psum[4][16][16];
  __shared__ float  z_sh[16][16];

  const int tid  = threadIdx.x;
  const int lane = tid & 31;
  const int wv   = tid >> 5;             // 0..3
  const int n0   = blockIdx.x * 16;
  const int row  = lane & 15;
  const int khalf = (lane >> 4) * 16;
  int bc = 0;

  for (int t = 0; t < TT; ++t) {
    // ---- stage current h (bf16) into LDS ----
#if USE_TDM
    if (wv == 0) {
      tdm_load_2d_bf16((unsigned int)(uintptr_t)&a_sh[0][0], hb,
                       HID, BB, HID, HID, BB);
      wait_tensor_le0();
    }
#else
    {
      const uint4* src = (const uint4*)hb;
      uint4* dst = (uint4*)&a_sh[0][0];
      for (int i = tid; i < (BB * HID * 2) / 16; i += 128) dst[i] = src[i];
    }
#endif
    __syncthreads();

    // phase A: z (waves 0,1) and r (waves 2,3), K split in halves
    {
      const __bf16* Wg = Whh + (size_t)(wv >> 1) * HID * HID;
      int cbeg = (wv & 1) * 16;
      v8f acc = {};
      for (int c = 0; c < 16; ++c) {
        int k0 = (cbeg + c) * 32;
        v16bf a = *(const v16bf*)&a_sh[row][k0 + khalf];
        v16bf b = *(const v16bf*)&Wg[(size_t)(n0 + row) * HID + k0 + khalf];
        acc = wmma_bf16(a, b, acc);
      }
      int m = (lane >> 4) * 8, n = lane & 15;
      for (int i = 0; i < 8; ++i) psum[wv][m + i][n] = acc[i];
    }
    __syncthreads();

    for (int e = tid; e < 256; e += 128) {
      int m = e >> 4, n = e & 15;
      size_t gxrow = ((size_t)t * BB + m) * G3;
      float zp = psum[0][m][n] + psum[1][m][n] + Gx[gxrow + 0 * HID + n0 + n];
      float rp = psum[2][m][n] + psum[3][m][n] + Gx[gxrow + 1 * HID + n0 + n];
      float zv = 1.0f / (1.0f + __expf(-zp));
      float rv = 1.0f / (1.0f + __expf(-rp));
      z_sh[m][n] = zv;
      float hv = hf[m * HID + n0 + n];
      rh[m * HID + n0 + n] = (__bf16)(rv * hv);
    }
    __threadfence();
    bc += 64;
    gbar(bar, bc);

    // ---- stage r*h into LDS ----
#if USE_TDM
    if (wv == 0) {
      tdm_load_2d_bf16((unsigned int)(uintptr_t)&a_sh[0][0], rh,
                       HID, BB, HID, HID, BB);
      wait_tensor_le0();
    }
#else
    {
      const uint4* src = (const uint4*)rh;
      uint4* dst = (uint4*)&a_sh[0][0];
      for (int i = tid; i < (BB * HID * 2) / 16; i += 128) dst[i] = src[i];
    }
#endif
    __syncthreads();

    // phase B: candidate, K split across 4 waves
    {
      const __bf16* Wc = Whh + (size_t)2 * HID * HID;
      v8f acc = {};
      for (int c = 0; c < 8; ++c) {
        int k0 = (wv * 8 + c) * 32;
        v16bf a = *(const v16bf*)&a_sh[row][k0 + khalf];
        v16bf b = *(const v16bf*)&Wc[(size_t)(n0 + row) * HID + k0 + khalf];
        acc = wmma_bf16(a, b, acc);
      }
      int m = (lane >> 4) * 8, n = lane & 15;
      for (int i = 0; i < 8; ++i) psum[wv][m + i][n] = acc[i];
    }
    __syncthreads();

    for (int e = tid; e < 256; e += 128) {
      int m = e >> 4, n = e & 15;
      size_t gxrow = ((size_t)t * BB + m) * G3;
      float cp = psum[0][m][n] + psum[1][m][n] + psum[2][m][n] + psum[3][m][n] +
                 Gx[gxrow + 2 * HID + n0 + n];
      float cv = tanhf(cp);
      float zv = z_sh[m][n];
      float hv = hf[m * HID + n0 + n];
      float hn = (1.0f - zv) * hv + zv * cv;
      hf[m * HID + n0 + n] = hn;
      hb[m * HID + n0 + n] = (__bf16)hn;
      out_h[((size_t)t * BB + m) * HID + n0 + n] = hn;
      hseq[((size_t)t * BB + m) * HID + n0 + n] = (__bf16)hn;
    }
    __threadfence();
    bc += 64;
    gbar(bar, bc);
  }
}

// ---------------- decode GEMM: all_y = hseq @ Wd^T ----------------
// [4096,1024] x [1024,32000]. Block tile 64x128, 8 waves (2x4), wave 32x32.
// TDM double-buffered: wave 0 DMAs next A(64x64)/B(128x64) bf16 tiles into
// LDS while all waves compute on the current stage.
#define BM 64
#define BN 128
#define BK 64
__global__ void __launch_bounds__(256) k_out_gemm(const __bf16* A,
                                                  const __bf16* Wd,
                                                  float* Y) {
  __shared__ __bf16 As[2][BM][BK];   // 2 x  8 KB
  __shared__ __bf16 Bs[2][BN][BK];   // 2 x 16 KB
  const int tid = threadIdx.x, lane = tid & 31, wv = tid >> 5;
  const int wm = wv >> 2, wn = wv & 3;              // 2 x 4 wave grid
  const int m0 = blockIdx.y * BM, n0 = blockIdx.x * BN;
  const int row = lane & 15, khalf = (lane >> 4) * 16;
  v8f acc[2][2] = {};

#if USE_TDM
  if (wv == 0) {   // prologue: DMA stage 0
    tdm_load_2d_bf16((unsigned int)(uintptr_t)&As[0][0][0],
                     &A[(size_t)m0 * HID], HID, MROWS, HID, BK, BM);
    tdm_load_2d_bf16((unsigned int)(uintptr_t)&Bs[0][0][0],
                     &Wd[(size_t)n0 * HID], HID, VOCAB, HID, BK, BN);
  }
#endif

  for (int kt = 0, it = 0; kt < HID; kt += BK, ++it) {
    const int cur = it & 1;
#if USE_TDM
    if (wv == 0) {
      if (kt + BK < HID) {
        const int nxt = cur ^ 1;
        tdm_load_2d_bf16((unsigned int)(uintptr_t)&As[nxt][0][0],
                         &A[(size_t)m0 * HID + kt + BK], HID, MROWS, HID, BK,
                         BM);
        tdm_load_2d_bf16((unsigned int)(uintptr_t)&Bs[nxt][0][0],
                         &Wd[(size_t)n0 * HID + kt + BK], HID, VOCAB, HID, BK,
                         BN);
        wait_tensor_le2();   // current stage (2 oldest ops) landed
      } else {
        wait_tensor_le0();
      }
    }
    __syncthreads();
#else
    if (kt + BK < HID) {
      __builtin_prefetch(&A[(size_t)(m0 + (tid >> 2)) * HID + kt + BK], 0, 1);
      __builtin_prefetch(&Wd[(size_t)(n0 + (tid >> 1)) * HID + kt + BK], 0, 1);
    }
    {
      int r = tid >> 2, c = (tid & 3) * 16;
      *(v16bf*)&As[cur][r][c] =
          *(const v16bf*)&A[(size_t)(m0 + r) * HID + kt + c];
    }
    for (int ch = tid; ch < 512; ch += 256) {
      int r = ch >> 2, c = (ch & 3) * 16;
      *(v16bf*)&Bs[cur][r][c] =
          *(const v16bf*)&Wd[(size_t)(n0 + r) * HID + kt + c];
    }
    __syncthreads();
#endif
    for (int kc = 0; kc < BK; kc += 32) {
      v16bf a0 = *(const v16bf*)&As[cur][wm * 32 + row][kc + khalf];
      v16bf a1 = *(const v16bf*)&As[cur][wm * 32 + 16 + row][kc + khalf];
      v16bf b0 = *(const v16bf*)&Bs[cur][wn * 32 + row][kc + khalf];
      v16bf b1 = *(const v16bf*)&Bs[cur][wn * 32 + 16 + row][kc + khalf];
      acc[0][0] = wmma_bf16(a0, b0, acc[0][0]);
      acc[0][1] = wmma_bf16(a0, b1, acc[0][1]);
      acc[1][0] = wmma_bf16(a1, b0, acc[1][0]);
      acc[1][1] = wmma_bf16(a1, b1, acc[1][1]);
    }
    __syncthreads();
  }

  for (int i2 = 0; i2 < 2; ++i2)
    for (int j2 = 0; j2 < 2; ++j2) {
      int mb = m0 + wm * 32 + i2 * 16 + (lane >> 4) * 8;
      int nb = n0 + wn * 32 + j2 * 16 + (lane & 15);
      for (int i = 0; i < 8; ++i)
        Y[(size_t)(mb + i) * VOCAB + nb] = acc[i2][j2][i];
    }
}

// ---------------- host entry ----------------
extern "C" void kernel_launch(void* const* d_in, const int* in_sizes, int n_in,
                              void* d_out, int out_size, void* d_ws,
                              size_t ws_size, hipStream_t stream) {
  const int*   x   = (const int*)d_in[0];
  const float* h0  = (const float*)d_in[1];
  const float* emb = (const float*)d_in[2];
  const float* Wz  = (const float*)d_in[3];
  const float* Wr  = (const float*)d_in[4];
  const float* W   = (const float*)d_in[5];
  const float* Wd  = (const float*)d_in[6];
  float* out   = (float*)d_out;                    // [all_h | all_y]
  float* out_y = out + (size_t)MROWS * HID;

  char* ws = (char*)d_ws;
  __bf16* xe   = (__bf16*)(ws + OFF_XE);
  __bf16* Whh  = (__bf16*)(ws + OFF_WHH);
  __bf16* Wxx  = (__bf16*)(ws + OFF_WXX);
  __bf16* Wdb  = (__bf16*)(ws + OFF_WD);
  float*  Gx   = (float*)(ws + OFF_GX);
  __bf16* hseq = (__bf16*)(ws + OFF_HSEQ);
  float*  hf   = (float*)(ws + OFF_HF);
  __bf16* hb   = (__bf16*)(ws + OFF_HB);
  __bf16* rh   = (__bf16*)(ws + OFF_RH);
  int*    bar  = (int*)(ws + OFF_BAR);

  k_init<<<64, 256, 0, stream>>>(h0, hf, hb, bar);
  k_gather<<<(MROWS * EMB + 255) / 256, 256, 0, stream>>>(x, emb, xe);
  k_conv_wh<<<(3 * HID * HID + 255) / 256, 256, 0, stream>>>(Wz, Wr, W, Whh);
  k_conv_wx<<<(3 * HID * EMB + 255) / 256, 256, 0, stream>>>(Wz, Wr, W, Wxx);
  k_conv_wd<<<(VOCAB * HID + 255) / 256, 256, 0, stream>>>(Wd, Wdb);

  k_gx_gemm<<<dim3(G3 / 128, MROWS / 16), 256, 0, stream>>>(xe, Wxx, Gx);

  k_gru_rec<<<HID / 16, 128, 0, stream>>>(Whh, Gx, hf, hb, rh, out, hseq, bar);

  k_out_gemm<<<dim3(VOCAB / BN, MROWS / BM), 256, 0, stream>>>(hseq, Wdb,
                                                              out_y);
}
